// SelfAttention1d_53154515256219
// MI455X (gfx1250) — matrix operand order repaired and checked by reading, not compiled, not run on hardware
//
#include <hip/hip_runtime.h>
#include <hip/hip_bf16.h>

#define B_ 8
#define C_ 256
#define L_ 2048
#define D_ 32   // CQK

typedef __attribute__((ext_vector_type(16))) _Float16 v16h;
typedef __attribute__((ext_vector_type(8)))  _Float16 v8h;
typedef __attribute__((ext_vector_type(8)))  float    v8f;

// ---------------------------------------------------------------------------
// Kernel 0: one-shot f32 -> f16 conversion of the (reused-everywhere) weights.
// ---------------------------------------------------------------------------
__global__ __launch_bounds__(256) void cvtw_kernel(
    const float* __restrict__ Wq, const float* __restrict__ Wk,
    const float* __restrict__ Wv,
    _Float16* __restrict__ wq16, _Float16* __restrict__ wk16,
    _Float16* __restrict__ wv16)
{
    const int i = blockIdx.x*256 + threadIdx.x;      // grid covers C_*C_
    if (i < D_*C_) {
        wq16[i] = (_Float16)Wq[i];
        wk16[i] = (_Float16)Wk[i];
    }
    wv16[i] = (_Float16)Wv[i];
}

// ---------------------------------------------------------------------------
// Kernel 1: 1x1-conv projections via WMMA.
// Block = 4 waves, handles (b, 32 positions). x tile staged transposed in LDS
// as xT[l][c] f16 so the A-operand (A[m=l][k=c]) is contiguous per lane.
// B-operand = f16 weight tile B[k=c][n=o]: two contiguous 16B loads of row
// w16[o] per K-chunk (lane-shared row, c-half selected by lane<16).
// D tile [M=l][N=o], accumulator initialized to bias[o] (per-lane splat).
// Outputs: q16,k16: [B][L][D] f16 ; v16: [B][C][L] f16.
// ---------------------------------------------------------------------------
__global__ __launch_bounds__(128) void proj_kernel(
    const float* __restrict__ x,
    const _Float16* __restrict__ wq16, const float* __restrict__ bq,
    const _Float16* __restrict__ wk16, const float* __restrict__ bk,
    const _Float16* __restrict__ wv16, const float* __restrict__ bv,
    _Float16* __restrict__ q16, _Float16* __restrict__ k16,
    _Float16* __restrict__ v16)
{
    __shared__ _Float16 xT[32][C_ + 8];       // [l_local][c], row = 528B (16B mult)
    const int t  = threadIdx.x;
    const int b  = blockIdx.x / (L_/32);
    const int l0 = (blockIdx.x % (L_/32)) * 32;
    const int wv   = t >> 5;
    const int lane = t & 31;

    // stage x[b, :, l0:l0+32] transposed (global reads coalesced along l)
    for (int idx = t; idx < C_*32; idx += 128) {
        const int c = idx >> 5, li = idx & 31;
        xT[li][c] = (_Float16)x[((size_t)b*C_ + c)*L_ + l0 + li];
    }
    __syncthreads();

    const int hi16  = lane >> 4;
    const int n     = lane & 15;
    const int koffA = hi16 ? 8 : 0;           // A-layout K offsets {koffA..+7, koffA+16..+23}
    const int coffB = hi16 ? 16 : 0;          // K offset of contiguous-16 order (unused here)
    const int rbase = hi16 ? 8 : 0;           // D-layout: VGPR r -> M = rbase + r

    // 40 D-tiles: [0,4) Q, [4,8) K, [8,40) V.  rel = (o_tile, l_subtile)
    for (int tid = wv; tid < 40; tid += 4) {
        const _Float16* Wsel;
        const float*    bsel;
        int rel;
        if (tid < 4)      { Wsel = wq16; bsel = bq; rel = tid;     }
        else if (tid < 8) { Wsel = wk16; bsel = bk; rel = tid - 4; }
        else              { Wsel = wv16; bsel = bv; rel = tid - 8; }
        const int ot = rel >> 1;              // o-tile
        const int lb = (rel & 1) * 16;        // l-subtile base
        const int og = ot*16 + n;             // this lane's output channel (N)

        v8f acc;
        {
            const float bias = bsel[og];
            #pragma unroll
            for (int r = 0; r < 8; ++r) acc[r] = bias;
        }

        const _Float16* xrow = &xT[lb + n][0];
        const _Float16* wrow = Wsel + (size_t)og*C_ + koffA;

        #pragma unroll
        for (int kc = 0; kc < 8; ++kc) {
            const int c0 = kc * 32;
            // A: x^T tile, contiguous LDS reads
            const v8h alo = *(const v8h*)(xrow + c0 + koffA);
            const v8h ahi = *(const v8h*)(xrow + c0 + koffA + 16);
            v16h aX;
            #pragma unroll
            for (int h = 0; h < 8; ++h) { aX[h] = alo[h]; aX[8+h] = ahi[h]; }
            // B: f16 weight row, two contiguous 16B loads
            const v8h wlo = *(const v8h*)(wrow + c0);
            const v8h whi = *(const v8h*)(wrow + c0 + 16);
            v16h bW;
            #pragma unroll
            for (int h = 0; h < 8; ++h) { bW[h] = wlo[h]; bW[8+h] = whi[h]; }

            acc = __builtin_amdgcn_wmma_f32_16x16x32_f16(false, aX, false, bW,
                                                         (short)0, acc, false, false);
        }

        if (tid < 8) {
            _Float16* dst = (tid < 4) ? q16 : k16;
            #pragma unroll
            for (int r = 0; r < 8; ++r) {
                const int ll = lb + rbase + r;
                dst[((size_t)b*L_ + l0 + ll)*D_ + og] = (_Float16)acc[r];
            }
        } else {
            #pragma unroll
            for (int r = 0; r < 8; ++r) {
                const int ll = lb + rbase + r;
                v16[((size_t)b*C_ + og)*L_ + l0 + ll] = (_Float16)acc[r];
            }
        }
    }
}

// ---------------------------------------------------------------------------
// Kernel 2: fused flash attention. One wave per (b, 16-query tile).
//   scores S = Q^T K  via V_WMMA_F32_16X16X32_F16 (K-dim == CQK == 32)
//   O      += V * P^T via V_WMMA_F32_16X16X32_F16 (K-dim == 32 j's per chunk)
// Per-lane softmax state (m,l) is owned by row i = lane%16.
// ---------------------------------------------------------------------------
__global__ __launch_bounds__(128) void attn_kernel(
    const float* __restrict__ x,
    const _Float16* __restrict__ q16,
    const _Float16* __restrict__ k16,
    const _Float16* __restrict__ v16,
    const float* __restrict__ gamma,
    float* __restrict__ out)
{
    __shared__ float sS[4][16*33];          // per-wave score bounce buffer
    const int wv   = threadIdx.x >> 5;
    const int lane = threadIdx.x & 31;
    const int tile = blockIdx.x*4 + wv;
    const int b    = tile / (L_/16);
    const int i0   = (tile % (L_/16)) * 16;

    const int hi16  = lane >> 4;            // 0: lanes 0-15, 1: lanes 16-31
    const int n     = lane & 15;            // N index / M index for A
    const int koffA = hi16 ? 8 : 0;         // A-layout K offsets: {koffA..+7, koffA+16..+23}
    const int coffB = hi16 ? 16 : 0;        // K offset of this code's contiguous-16 order (V, P)
    const int rbase = hi16 ? 8 : 0;         // D-layout: VGPR r -> M = rbase + r
    const int jb    = coffB;                // this lane's j-half within the 32-chunk

    // resident A operand: Q tile, A[m=i][k=c], q16 row contiguous in c
    v16h aQ;
    {
        const _Float16* qrow = q16 + ((size_t)b*L_ + i0 + n)*D_;
        const v8h lo = *(const v8h*)(qrow + koffA);
        const v8h hi = *(const v8h*)(qrow + koffA + 16);
        #pragma unroll
        for (int h = 0; h < 8; ++h) { aQ[h] = lo[h]; aQ[8+h] = hi[h]; }
    }

    const v8f zero = {0.f,0.f,0.f,0.f,0.f,0.f,0.f,0.f};
    v8f O[16];
    #pragma unroll
    for (int tt = 0; tt < 16; ++tt) O[tt] = zero;
    float m = -INFINITY, l = 0.f;

    float* sp = &sS[wv][0];

    for (int j0 = 0; j0 < L_; j0 += 32) {
        // ---- scores: two 16x16 tiles over j ----
        v16h bK0, bK1;
        {
            const _Float16* kr0 = k16 + ((size_t)b*L_ + j0 + n)*D_ + koffA;
            const _Float16* kr1 = k16 + ((size_t)b*L_ + j0 + 16 + n)*D_ + koffA;
            const v8h a0 = *(const v8h*)kr0, a1 = *(const v8h*)(kr0 + 16);
            const v8h b0 = *(const v8h*)kr1, b1 = *(const v8h*)(kr1 + 16);
            #pragma unroll
            for (int h = 0; h < 8; ++h) {
                bK0[h] = a0[h]; bK0[8+h] = a1[h];
                bK1[h] = b0[h]; bK1[8+h] = b1[h];
            }
        }
        v8f s0 = __builtin_amdgcn_wmma_f32_16x16x32_f16(false, aQ, false, bK0,
                                                        (short)0, zero, false, false);
        v8f s1 = __builtin_amdgcn_wmma_f32_16x16x32_f16(false, aQ, false, bK1,
                                                        (short)0, zero, false, false);

        // ---- D-layout -> row-major via LDS (per-wave private region) ----
        #pragma unroll
        for (int r = 0; r < 8; ++r) {
            sp[(rbase + r)*33 + n]      = s0[r];
            sp[(rbase + r)*33 + 16 + n] = s1[r];
        }
        asm volatile("s_wait_dscnt 0" ::: "memory");

        // lane owns row i = n; this lane handles j-half [jb, jb+16)
        float rv[16];
        #pragma unroll
        for (int h = 0; h < 16; ++h) rv[h] = sp[n*33 + jb + h];

        float cm = rv[0];
        #pragma unroll
        for (int h = 1; h < 16; ++h) cm = fmaxf(cm, rv[h]);
        cm = fmaxf(cm, __shfl_xor(cm, 16, 32));          // full-row max

        const float mn   = fmaxf(m, cm);
        const float corr = __expf(m - mn);

        // P in B-layout: half h <-> j = jb + h, N = i = n  (exactly this lane's half)
        v16h pB;
        float ls = 0.f;
        #pragma unroll
        for (int h = 0; h < 16; ++h) {
            const float p = __expf(rv[h] - mn);
            ls += p;
            pB[h] = (_Float16)p;
        }
        ls += __shfl_xor(ls, 16, 32);
        l = l*corr + ls;
        m = mn;

        // ---- O = O*corr + V * P^T, 16 channel tiles ----
        #pragma unroll
        for (int tt = 0; tt < 16; ++tt) {
            const _Float16* vrow = v16 + ((size_t)b*C_ + tt*16 + n)*L_ + j0 + coffB;
            const v8h lo = *(const v8h*)vrow;
            const v8h hi = *(const v8h*)(vrow + 8);
            v16h aV;
            #pragma unroll
            for (int h = 0; h < 8; ++h) { aV[h] = lo[h]; aV[8+h] = hi[h]; }
            O[tt] = O[tt] * corr;
            O[tt] = __builtin_amdgcn_wmma_f32_16x16x32_f16(false, aV, false, pB,
                                                           (short)0, O[tt], false, false);
        }
    }

    // ---- epilogue: out = gamma * O/l + x ----
    const float g    = gamma[0];
    const float invl = 1.f / l;
    #pragma unroll
    for (int tt = 0; tt < 16; ++tt) {
        #pragma unroll
        for (int r = 0; r < 8; ++r) {
            const int c = tt*16 + rbase + r;
            const size_t idx = ((size_t)b*C_ + c)*L_ + i0 + n;
            out[idx] = g * O[tt][r] * invl + x[idx];
        }
    }
}

extern "C" void kernel_launch(void* const* d_in, const int* in_sizes, int n_in,
                              void* d_out, int out_size, void* d_ws, size_t ws_size,
                              hipStream_t stream) {
    (void)in_sizes; (void)n_in; (void)out_size; (void)ws_size;
    const float* x     = (const float*)d_in[0];
    const float* Wq    = (const float*)d_in[1];
    const float* bq    = (const float*)d_in[2];
    const float* Wk    = (const float*)d_in[3];
    const float* bk    = (const float*)d_in[4];
    const float* Wv    = (const float*)d_in[5];
    const float* bv    = (const float*)d_in[6];
    const float* gamma = (const float*)d_in[7];
    float* out = (float*)d_out;

    _Float16* q16  = (_Float16*)d_ws;                      // B*L*D halfs = 1 MB
    _Float16* k16  = q16 + (size_t)B_*L_*D_;               // 1 MB
    _Float16* v16  = k16 + (size_t)B_*L_*D_;               // B*C*L halfs = 8 MB
    _Float16* wq16 = v16 + (size_t)B_*C_*L_;               // 16 KB
    _Float16* wk16 = wq16 + (size_t)D_*C_;                 // 16 KB
    _Float16* wv16 = wk16 + (size_t)D_*C_;                 // 128 KB

    cvtw_kernel<<<(C_*C_)/256, 256, 0, stream>>>(Wq, Wk, Wv, wq16, wk16, wv16);
    proj_kernel<<<B_*(L_/32), 128, 0, stream>>>(x, wq16, bq, wk16, bk, wv16, bv,
                                                q16, k16, v16);
    attn_kernel<<<(B_*(L_/16))/4, 128, 0, stream>>>(x, q16, k16, v16, gamma, out);
}
